// AttentionBlock_90168543412425
// MI455X (gfx1250) — matrix operand order repaired and checked by reading, not compiled, not run on hardware
//
#include <hip/hip_runtime.h>

// ---------------------------------------------------------------------------
// AttentionBlock for MI455X (gfx1250, wave32).
// B=32, N=8192, E=A=128.  Core GEMMs run on v_wmma_f32_16x16x32_f16.
// Bandwidth-bound: ~134MB of embeddings, L2-resident (192MB) across passes.
// Deterministic: no atomics; context uses fixed-order partial reduction.
// v2: A-strip fully prefetched (one 16x global_load_b128 clause, no per-kstep
//     loadcnt stalls); 256 rows per block to halve W LDS-staging traffic.
// ---------------------------------------------------------------------------

typedef __attribute__((ext_vector_type(16))) _Float16 v16h;
typedef __attribute__((ext_vector_type(8)))  _Float16 v8h;
typedef __attribute__((ext_vector_type(8)))  float    v8f;

#define BB 32
#define NN 8192
#define EE 128
#define AA 128
#define LDS_STRIDE 136   // halves; 272B rows -> 16B-aligned, banks spread

// ---------------------------------------------------------------------------
// Kernel 1: convert first halves of W_a / W_o (the emb-facing columns) to f16.
__global__ __launch_bounds__(256) void ab_prep_w(
    const float* __restrict__ Wa, const float* __restrict__ Wo,
    _Float16* __restrict__ w16a, _Float16* __restrict__ w16o) {
  int idx = blockIdx.x * 256 + threadIdx.x;            // 0..16383
  int src = ((idx >> 7) << 8) | (idx & 127);           // row*256 + col
  w16a[idx] = (_Float16)Wa[src];
  w16o[idx] = (_Float16)Wo[src];
}

// ---------------------------------------------------------------------------
// Kernel 2: bias_a[b][a] = sum_e W_a[a][128+e] * gru[b][e]
__global__ __launch_bounds__(128) void ab_bias_a(
    const float* __restrict__ Wa, const float* __restrict__ gru,
    float* __restrict__ biasa) {
  int b = blockIdx.x, a = threadIdx.x;
  const float* wr = Wa + a * 256 + 128;
  const float* g  = gru + b * EE;
  float acc = 0.f;
#pragma unroll 8
  for (int e = 0; e < EE; ++e) acc += wr[e] * g[e];
  biasa[b * AA + a] = acc;
}

// ---------------------------------------------------------------------------
// Kernel 3 (the WMMA workhorse, used twice):
// scores[b,n] = v . relu( emb[b,n,:] @ W16^T + bias[b,:] )
// Block: b = blockIdx.y, 256 rows of n per block; 8 waves x (2 x 16 rows).
__global__ __launch_bounds__(256) void ab_scores(
    const float* __restrict__ emb, const _Float16* __restrict__ W16,
    const float* __restrict__ bias, const float* __restrict__ v,
    float* __restrict__ scores) {
  __shared__ __attribute__((aligned(32))) _Float16 ldsW[128 * LDS_STRIDE];
  __shared__ float ldsBias[AA];
  __shared__ float ldsV[AA];

  const int tid = threadIdx.x;
  const int b   = blockIdx.y;

  // Stage W (f16, row-major [A=128, E=128]) into padded LDS; copy as dwords.
  {
    const unsigned int* wg = (const unsigned int*)W16;     // 8192 dwords
    unsigned int* wl = (unsigned int*)ldsW;                // stride 68 dwords
    for (int i = tid; i < 128 * 64; i += 256) {
      int row = i >> 6, col = i & 63;
      wl[row * (LDS_STRIDE / 2) + col] = wg[i];
    }
  }
  if (tid < AA) { ldsBias[tid] = bias[b * AA + tid]; ldsV[tid] = v[tid]; }
  __syncthreads();

  const int wave = tid >> 5;
  const int lane = tid & 31;
  const int l15  = lane & 15;
  const int hl   = lane >> 4;                 // half-wave select

#pragma unroll 1
  for (int sub = 0; sub < 2; ++sub) {         // two 16-row tiles per wave
    const int nbase = blockIdx.x * 256 + sub * 128 + wave * 16;
    const float* row = emb + ((size_t)b * NN + nbase + l15) * EE;

    // --- Prefetch the lane's whole A strip (64 floats) in one clause.
    // Lanes 0-15 provide K = {k0..k0+7, k0+16..k0+23} per kstep,
    // lanes 16-31 provide K = {k0+8..k0+15, k0+24..k0+31}  (ISA A layout).
    float4 raw[16];
#pragma unroll
    for (int ks = 0; ks < 4; ++ks) {
      const float4* p0 = (const float4*)(row + ks * 32 + hl * 8);
      const float4* p1 = (const float4*)(row + ks * 32 + hl * 8 + 16);
      raw[ks * 4 + 0] = p0[0];
      raw[ks * 4 + 1] = p0[1];
      raw[ks * 4 + 2] = p1[0];
      raw[ks * 4 + 3] = p1[1];
    }

    // --- Convert once into 4 persistent f16 A-fragments.
    v16h af[4];
#pragma unroll
    for (int ks = 0; ks < 4; ++ks) {
      float4 x0 = raw[ks * 4 + 0], x1 = raw[ks * 4 + 1];
      float4 y0 = raw[ks * 4 + 2], y1 = raw[ks * 4 + 3];
      v16h a;
      a[0] = (_Float16)x0.x; a[1] = (_Float16)x0.y; a[2] = (_Float16)x0.z; a[3] = (_Float16)x0.w;
      a[4] = (_Float16)x1.x; a[5] = (_Float16)x1.y; a[6] = (_Float16)x1.z; a[7] = (_Float16)x1.w;
      a[8] = (_Float16)y0.x; a[9] = (_Float16)y0.y; a[10] = (_Float16)y0.z; a[11] = (_Float16)y0.w;
      a[12] = (_Float16)y1.x; a[13] = (_Float16)y1.y; a[14] = (_Float16)y1.z; a[15] = (_Float16)y1.w;
      af[ks] = a;
    }

    // --- 32 WMMAs: B fragments streamed from LDS.
    v8f acc[8] = {};
#pragma unroll
    for (int ks = 0; ks < 4; ++ks) {
      const int kB = ks * 32 + hl * 8;        // column N=l15(+16t), K = {kB..kB+7, kB+16..kB+23}
#pragma unroll
      for (int t = 0; t < 8; ++t) {
        const _Float16* bp = &ldsW[(t * 16 + l15) * LDS_STRIDE + kB];
        v8h b0 = *(const v8h*)bp;
        v8h b1 = *(const v8h*)(bp + 16);
        v16h bf;
#pragma unroll
        for (int q = 0; q < 8; ++q) { bf[q] = b0[q]; bf[8 + q] = b1[q]; }
        acc[t] = __builtin_amdgcn_wmma_f32_16x16x32_f16(
            false, af[ks], false, bf, (short)0, acc[t], false, false);
      }
    }

    // --- Epilogue: +bias, relu, *v, then sum over the 128 'a' columns.
    float rowsum[8] = {0.f, 0.f, 0.f, 0.f, 0.f, 0.f, 0.f, 0.f};
#pragma unroll
    for (int t = 0; t < 8; ++t) {
      float bt = ldsBias[t * 16 + l15];
      float vt = ldsV[t * 16 + l15];
#pragma unroll
      for (int j = 0; j < 8; ++j) {
        float s = acc[t][j] + bt;
        s = s > 0.f ? s : 0.f;
        rowsum[j] += s * vt;
      }
    }
#pragma unroll
    for (int j = 0; j < 8; ++j) {
      float s = rowsum[j];
      s += __shfl_xor(s, 1, 32);
      s += __shfl_xor(s, 2, 32);
      s += __shfl_xor(s, 4, 32);
      s += __shfl_xor(s, 8, 32);             // reduce the 16-lane column group
      if (l15 == 0) {
        int m = nbase + j + hl * 8;          // C-layout row M = j + 8*half
        scores[(size_t)b * NN + m] = s;
      }
    }
  }
}

// ---------------------------------------------------------------------------
// Kernel 4: per-batch softmax stats over scores_a (max and 1/sum(exp)).
__global__ __launch_bounds__(256) void ab_stats(
    const float* __restrict__ sa, float* __restrict__ st) {
  __shared__ float red[256];
  int b = blockIdx.x, tid = threadIdx.x;
  const float* p = sa + (size_t)b * NN;
  float m = -INFINITY;
  for (int i = tid; i < NN; i += 256) m = fmaxf(m, p[i]);
  red[tid] = m; __syncthreads();
  for (int s = 128; s > 0; s >>= 1) {
    if (tid < s) red[tid] = fmaxf(red[tid], red[tid + s]);
    __syncthreads();
  }
  float mx = red[0]; __syncthreads();
  float sum = 0.f;
  for (int i = tid; i < NN; i += 256) sum += __expf(p[i] - mx);
  red[tid] = sum; __syncthreads();
  for (int s = 128; s > 0; s >>= 1) {
    if (tid < s) red[tid] += red[tid + s];
    __syncthreads();
  }
  if (tid == 0) { st[b * 2] = mx; st[b * 2 + 1] = 1.f / red[0]; }
}

// ---------------------------------------------------------------------------
// Kernel 5: context partials.  part[b][chunk][e] = sum over 512 n of w*emb.
__global__ __launch_bounds__(128) void ab_context(
    const float* __restrict__ emb, const float* __restrict__ sa,
    const float* __restrict__ st, float* __restrict__ part) {
  __shared__ float w[512];
  int ch = blockIdx.x, b = blockIdx.y, tid = threadIdx.x;
  int n0 = ch * 512;
  float mx = st[b * 2], inv = st[b * 2 + 1];
  const float* p = sa + (size_t)b * NN + n0;
  for (int i = tid; i < 512; i += 128) w[i] = __expf(p[i] - mx) * inv;
  __syncthreads();
  const float* eb = emb + ((size_t)b * NN + n0) * EE + tid;
  float acc = 0.f;
#pragma unroll 8
  for (int i = 0; i < 512; ++i) acc += w[i] * eb[(size_t)i * EE];
  part[(b * 16 + ch) * EE + tid] = acc;
}

// ---------------------------------------------------------------------------
// Kernel 6: c[b] = fixed-order sum of partials; bias_o[b][a] = W_o[a,128:]@c.
__global__ __launch_bounds__(128) void ab_bias_o(
    const float* __restrict__ part, const float* __restrict__ Wo,
    float* __restrict__ biaso) {
  __shared__ float c[EE];
  int b = blockIdx.x, tid = threadIdx.x;
  float s = 0.f;
#pragma unroll
  for (int ch = 0; ch < 16; ++ch) s += part[(b * 16 + ch) * EE + tid];
  c[tid] = s; __syncthreads();
  const float* wr = Wo + tid * 256 + 128;
  float acc = 0.f;
#pragma unroll 8
  for (int e = 0; e < EE; ++e) acc += wr[e] * c[e];
  biaso[b * AA + tid] = acc;
}

// ---------------------------------------------------------------------------
// Kernel 7: final masked softmax over o_score -> d_out.
__global__ __launch_bounds__(256) void ab_softmax(
    const float* __restrict__ so, const float* __restrict__ mask,
    float* __restrict__ out) {
  __shared__ float x[NN];
  __shared__ float red[256];
  int b = blockIdx.x, tid = threadIdx.x;
  const float* p = so + (size_t)b * NN;
  const float* mk = mask + (size_t)b * NN;
  float m = -INFINITY;
  for (int i = tid; i < NN; i += 256) {
    float v = p[i] + mk[i];
    x[i] = v;
    m = fmaxf(m, v);
  }
  red[tid] = m; __syncthreads();
  for (int s = 128; s > 0; s >>= 1) {
    if (tid < s) red[tid] = fmaxf(red[tid], red[tid + s]);
    __syncthreads();
  }
  float mx = red[0]; __syncthreads();
  float sum = 0.f;
  for (int i = tid; i < NN; i += 256) {
    float e = __expf(x[i] - mx);
    x[i] = e;
    sum += e;
  }
  red[tid] = sum; __syncthreads();
  for (int s = 128; s > 0; s >>= 1) {
    if (tid < s) red[tid] += red[tid + s];
    __syncthreads();
  }
  float inv = 1.f / red[0];
  float* o = out + (size_t)b * NN;
  for (int i = tid; i < NN; i += 256) o[i] = x[i] * inv;
}

// ---------------------------------------------------------------------------
extern "C" void kernel_launch(void* const* d_in, const int* in_sizes, int n_in,
                              void* d_out, int out_size, void* d_ws, size_t ws_size,
                              hipStream_t stream) {
  const float* emb  = (const float*)d_in[0];   // [32, 8192, 128]
  const float* gru  = (const float*)d_in[1];   // [32, 1, 128]
  const float* mask = (const float*)d_in[2];   // [32, 8192]
  const float* v_a  = (const float*)d_in[3];   // [128]
  const float* W_a  = (const float*)d_in[4];   // [128, 256]
  const float* v_o  = (const float*)d_in[5];   // [128]
  const float* W_o  = (const float*)d_in[6];   // [128, 256]
  float* out = (float*)d_out;                  // [32, 8192]

  // Workspace layout (~2.5 MB total)
  char* ws = (char*)d_ws;
  _Float16* w16a = (_Float16*)(ws + 0);                 // 32 KB
  _Float16* w16o = (_Float16*)(ws + 32768);             // 32 KB
  float* biasa = (float*)(ws + 65536);                  // 16 KB
  float* biaso = (float*)(ws + 81920);                  // 16 KB
  float* sa    = (float*)(ws + 98304);                  // 1 MB
  float* so    = (float*)(ws + 98304 + 1048576);        // 1 MB
  float* st    = (float*)(ws + 98304 + 2097152);        // 256 B
  float* part  = (float*)(ws + 98304 + 2097152 + 256);  // 256 KB

  ab_prep_w<<<64, 256, 0, stream>>>(W_a, W_o, w16a, w16o);
  ab_bias_a<<<BB, 128, 0, stream>>>(W_a, gru, biasa);
  ab_scores<<<dim3(NN / 256, BB), 256, 0, stream>>>(emb, w16a, biasa, v_a, sa);
  ab_stats<<<BB, 256, 0, stream>>>(sa, st);
  ab_context<<<dim3(16, BB), 128, 0, stream>>>(emb, sa, st, part);
  ab_bias_o<<<BB, 128, 0, stream>>>(part, W_o, biaso);
  ab_scores<<<dim3(NN / 256, BB), 256, 0, stream>>>(emb, w16o, biaso, v_o, so);
  ab_softmax<<<BB, 256, 0, stream>>>(so, mask, out);
}
